// SelfAttention_73091753443697
// MI455X (gfx1250) — matrix operand order repaired and checked by reading, not compiled, not run on hardware
//
#include <hip/hip_runtime.h>
#include <hip/hip_bf16.h>

typedef _Float16 half_t;
typedef __attribute__((ext_vector_type(16))) _Float16 v16h;
typedef __attribute__((ext_vector_type(8)))  _Float16 v8h;
typedef __attribute__((ext_vector_type(4)))  _Float16 v4h;
typedef __attribute__((ext_vector_type(8)))  float    v8f;

union V16H { v16h v; v8h h[2]; };

#define H_DIM   768
#define NHEADS  12
#define DHEAD   64
#define SEQ     2048
#define BATCH   4
#define ROWS    (BATCH * SEQ)   // 8192
#define HCAT    (3 * H_DIM)     // 2304

__device__ __forceinline__ v8f wmma_f16(v16h a, v16h b, v8f c) {
  return __builtin_amdgcn_wmma_f32_16x16x32_f16(false, a, false, b, (short)0, c, false, false);
}

// A-fragment: 16x32 f16 tile, row-major source. lane m = lane&15, half g = lane>>4.
// elems 0..7  -> k = k0 + g*8 ; elems 8..15 -> k = k0 + 16 + g*8
__device__ __forceinline__ v16h load_frag_a(const half_t* __restrict__ base, int ld,
                                            int row0, int k0) {
  const unsigned lane = threadIdx.x & 31u;
  const int m = (int)(lane & 15u), g = (int)(lane >> 4u);
  const half_t* p = base + (size_t)(row0 + m) * ld + k0 + g * 8;
  V16H r;
  r.h[0] = *(const v8h*)(p);
  r.h[1] = *(const v8h*)(p + 16);
  return r.v;
}

// B-fragment: 32x16 f16 tile; source stored "transposed" (row index = output column n).
// lane n = lane&15, half g = lane>>4; elems 0..7 -> k = k0 + g*8 ; elems 8..15 -> k = k0 + 16 + g*8
__device__ __forceinline__ v16h load_frag_b(const half_t* __restrict__ base, int ld,
                                            int col0, int k0) {
  const unsigned lane = threadIdx.x & 31u;
  const int n = (int)(lane & 15u), g = (int)(lane >> 4u);
  const half_t* p = base + (size_t)(col0 + n) * ld + k0 + g * 8;
  V16H r;
  r.h[0] = *(const v8h*)(p);
  r.h[1] = *(const v8h*)(p + 16);
  return r.v;
}

// ---------------- conversion kernels ----------------

__global__ __launch_bounds__(256) void cvt_f16_kernel(const float4* __restrict__ in,
                                                      half_t* __restrict__ out, int n4) {
  int i = blockIdx.x * 256 + threadIdx.x;
  if (i < n4) {
    float4 f = in[i];
    v4h h;
    h.x = (half_t)f.x; h.y = (half_t)f.y; h.z = (half_t)f.z; h.w = (half_t)f.w;
    *(v4h*)(out + 4 * (size_t)i) = h;
  }
}

// Wt[o*N + i] = (f16) W[i*N + o]   (LDS-tiled, coalesced both sides)
__global__ __launch_bounds__(256) void transpose_cvt_kernel(const float* __restrict__ W,
                                                            half_t* __restrict__ Wt, int N) {
  __shared__ half_t tile[16][17];
  const int bx = blockIdx.x * 16, by = blockIdx.y * 16;
  const int tx = (int)(threadIdx.x & 15u), ty = (int)(threadIdx.x >> 4u);
  tile[ty][tx] = (half_t)W[(size_t)(by + ty) * N + bx + tx];
  __syncthreads();
  Wt[(size_t)(bx + ty) * N + by + tx] = tile[tx][ty];
}

// ---------------- fused QKV projection GEMM ----------------
// X:[8192x768] f16, Wcat:[2304x768] f16 (rows 0..767 = Q out-cols, then K, then V).
// Each wave: 32x64 output tile (2 A-frags share 4 B-frags -> 8 WMMA / 6 frag loads).
__global__ __launch_bounds__(256) void gemm_qkv_kernel(const half_t* __restrict__ X,
                                                       const half_t* __restrict__ Wcat,
                                                       const float* __restrict__ bq,
                                                       const float* __restrict__ bk,
                                                       const float* __restrict__ bv,
                                                       half_t* __restrict__ Qh,
                                                       half_t* __restrict__ Kh,
                                                       half_t* __restrict__ Vth) {
  const unsigned tid = threadIdx.x;
  const unsigned wave = tid >> 5, lane = tid & 31u;
  const int wm = (int)(wave & 3u), wn = (int)(wave >> 2u);
  const int row0 = blockIdx.x * 128 + wm * 32;          // 32 rows / wave
  const int colBase = blockIdx.y * 128 + wn * 64;       // 64 cols / wave (within one region)

  v8f acc0[4] = {v8f{}, v8f{}, v8f{}, v8f{}};
  v8f acc1[4] = {v8f{}, v8f{}, v8f{}, v8f{}};
#pragma unroll 2
  for (int k0 = 0; k0 < H_DIM; k0 += 32) {
    v16h a0 = load_frag_a(X, H_DIM, row0, k0);
    v16h a1 = load_frag_a(X, H_DIM, row0 + 16, k0);
#pragma unroll
    for (int t = 0; t < 4; ++t) {
      v16h b = load_frag_b(Wcat, H_DIM, colBase + t * 16, k0);
      acc0[t] = wmma_f16(a0, b, acc0[t]);
      acc1[t] = wmma_f16(a1, b, acc1[t]);
    }
  }

  const int region = colBase / H_DIM;                   // 0=Q 1=K 2=V (uniform per wave)
  const int cbase = colBase - region * H_DIM;
  const float* bias = (region == 0) ? bq : (region == 1) ? bk : bv;
  half_t* out = (region == 0) ? Qh : (region == 1) ? Kh : Vth;
  const float scale = (region == 0) ? 0.125f : 1.0f;    // fold 1/sqrt(64) into Q

  const int n = (int)(lane & 15u), g = (int)(lane >> 4u);
#pragma unroll
  for (int sub = 0; sub < 2; ++sub) {
    const v8f* acc = sub ? acc1 : acc0;
#pragma unroll
    for (int t = 0; t < 4; ++t) {
#pragma unroll
      for (int r = 0; r < 8; ++r) {
        const int rr = row0 + sub * 16 + r + 8 * g;     // b*2048+s
        const int cc = cbase + t * 16 + n;              // h*64+d
        const float v = (acc[t][r] + bias[cc]) * scale;
        const int b = rr >> 11, s = rr & 2047;
        const int h = cc >> 6,  d = cc & 63;
        size_t idx;
        if (region == 2) idx = ((size_t)(b * NHEADS + h) * DHEAD + d) * SEQ + s;  // V^T
        else             idx = ((size_t)(b * NHEADS + h) * SEQ + s) * DHEAD + d;  // Q,K
        out[idx] = (half_t)v;
      }
    }
  }
}

// ---------------- fused flash attention ----------------
// Q,K: [B,NH,S,64] f16 (Q pre-scaled by 0.125), Vt: [B,NH,64,S] f16
// Aout: [B*S, 768] f16 (heads merged back). Mask ignored: the reference broadcasts
// it onto the query axis, which is softmax-shift-invariant.
__global__ __launch_bounds__(256) void attn_kernel(const half_t* __restrict__ Q,
                                                   const half_t* __restrict__ K,
                                                   const half_t* __restrict__ Vt,
                                                   half_t* __restrict__ Aout) {
  __shared__ half_t Pl[8][16 * 72];   // per-wave P staging, padded row stride 72
  const unsigned tid = threadIdx.x;
  const unsigned wave = tid >> 5, lane = tid & 31u;
  const int n = (int)(lane & 15u), g = (int)(lane >> 4u);
  const int bh = blockIdx.y;                         // b*NH + h
  const int q0 = blockIdx.x * 128 + (int)wave * 16;  // 16 q-rows per wave

  const half_t* Qh = Q  + (size_t)bh * SEQ * DHEAD;
  const half_t* Kh = K  + (size_t)bh * SEQ * DHEAD;
  const half_t* Vh = Vt + (size_t)bh * DHEAD * SEQ;
  half_t* P = &Pl[wave][0];

  v16h qa0 = load_frag_a(Qh, DHEAD, q0, 0);
  v16h qa1 = load_frag_a(Qh, DHEAD, q0, 32);

  float rmax[8], rsum[8];
#pragma unroll
  for (int r = 0; r < 8; ++r) { rmax[r] = -1e30f; rsum[r] = 0.0f; }
  v8f o[4] = {v8f{}, v8f{}, v8f{}, v8f{}};

  for (int kb = 0; kb < SEQ; kb += 64) {
    // ---- scores: 16 x 64 tile (4 x 16x16 WMMA tiles, K=64 in 2 chunks) ----
    v8f s[4] = {v8f{}, v8f{}, v8f{}, v8f{}};
#pragma unroll
    for (int j = 0; j < 4; ++j) {
      v16h b0 = load_frag_b(Kh, DHEAD, kb + j * 16, 0);
      s[j] = wmma_f16(qa0, b0, s[j]);
      v16h b1 = load_frag_b(Kh, DHEAD, kb + j * 16, 32);
      s[j] = wmma_f16(qa1, b1, s[j]);
    }
    // ---- issue V-fragment loads NOW; latency hides under softmax VALU below ----
    v16h vb[8];
#pragma unroll
    for (int t = 0; t < 4; ++t) {
      vb[2 * t]     = load_frag_b(Vh, SEQ, t * 16, kb);
      vb[2 * t + 1] = load_frag_b(Vh, SEQ, t * 16, kb + 32);
    }
    // prefetch next key block into cache while we do softmax
    if (kb + 64 < SEQ) {
      __builtin_prefetch(Kh + (size_t)(kb + 64 + (int)lane) * DHEAD, 0, 1);
      __builtin_prefetch(Vh + (size_t)((int)lane) * SEQ + kb + 64, 0, 1);
    }
    // ---- online softmax: row max over 64 keys (4 tiles x 16 lanes) ----
    float alpha[8];
#pragma unroll
    for (int r = 0; r < 8; ++r) {
      float v = fmaxf(fmaxf(s[0][r], s[1][r]), fmaxf(s[2][r], s[3][r]));
      v = fmaxf(v, __shfl_xor(v, 1, 32));
      v = fmaxf(v, __shfl_xor(v, 2, 32));
      v = fmaxf(v, __shfl_xor(v, 4, 32));
      v = fmaxf(v, __shfl_xor(v, 8, 32));
      const float mnew = fmaxf(rmax[r], v);
      alpha[r] = __expf(rmax[r] - mnew);
      rmax[r] = mnew;
    }
    // ---- P = exp(s - m), stage f16 tile to LDS (C-layout -> A-layout via LDS) ----
#pragma unroll
    for (int j = 0; j < 4; ++j) {
#pragma unroll
      for (int r = 0; r < 8; ++r) {
        const float p = __expf(s[j][r] - rmax[r]);
        s[j][r] = p;
        P[(r + 8 * g) * 72 + j * 16 + n] = (half_t)p;
      }
    }
#pragma unroll
    for (int r = 0; r < 8; ++r) {
      float v = (s[0][r] + s[1][r]) + (s[2][r] + s[3][r]);
      v += __shfl_xor(v, 1, 32);
      v += __shfl_xor(v, 2, 32);
      v += __shfl_xor(v, 4, 32);
      v += __shfl_xor(v, 8, 32);
      rsum[r] = rsum[r] * alpha[r] + v;
#pragma unroll
      for (int t = 0; t < 4; ++t) o[t][r] *= alpha[r];
    }
    asm volatile("s_wait_dscnt 0" ::: "memory");   // wave-local P staging visible
    // ---- reload P as A-fragments (two K=32 chunks) ----
    V16H pa0, pa1;
    {
      const half_t* pp = P + n * 72 + g * 8;       // A-frag: row m = lane&15
      pa0.h[0] = *(const v8h*)(pp);
      pa0.h[1] = *(const v8h*)(pp + 16);
      pa1.h[0] = *(const v8h*)(pp + 32);
      pa1.h[1] = *(const v8h*)(pp + 48);
    }
    // ---- O += P @ V  (V fragments already in flight/registers) ----
#pragma unroll
    for (int t = 0; t < 4; ++t) {
      o[t] = wmma_f16(pa0.v, vb[2 * t], o[t]);
      o[t] = wmma_f16(pa1.v, vb[2 * t + 1], o[t]);
    }
  }

  // ---- normalize and write merged-head f16 output [B*S, 768] ----
  const int b = bh / NHEADS, h = bh % NHEADS;
#pragma unroll
  for (int r = 0; r < 8; ++r) {
    const float rinv = 1.0f / rsum[r];
    const int srow = q0 + r + 8 * g;
    half_t* orow = Aout + (size_t)(b * SEQ + srow) * H_DIM + h * DHEAD;
#pragma unroll
    for (int t = 0; t < 4; ++t) orow[t * 16 + n] = (half_t)(o[t][r] * rinv);
  }
}

// ---------------- output projection GEMM (f32 out) ----------------
__global__ __launch_bounds__(256) void gemm_out_kernel(const half_t* __restrict__ A,
                                                       const half_t* __restrict__ Wt,
                                                       const float* __restrict__ bias,
                                                       float* __restrict__ out) {
  const unsigned tid = threadIdx.x;
  const unsigned wave = tid >> 5, lane = tid & 31u;
  const int wm = (int)(wave & 3u), wn = (int)(wave >> 2u);
  const int row0 = blockIdx.x * 128 + wm * 32;
  const int colBase = blockIdx.y * 128 + wn * 64;

  v8f acc0[4] = {v8f{}, v8f{}, v8f{}, v8f{}};
  v8f acc1[4] = {v8f{}, v8f{}, v8f{}, v8f{}};
#pragma unroll 2
  for (int k0 = 0; k0 < H_DIM; k0 += 32) {
    v16h a0 = load_frag_a(A, H_DIM, row0, k0);
    v16h a1 = load_frag_a(A, H_DIM, row0 + 16, k0);
#pragma unroll
    for (int t = 0; t < 4; ++t) {
      v16h b = load_frag_b(Wt, H_DIM, colBase + t * 16, k0);
      acc0[t] = wmma_f16(a0, b, acc0[t]);
      acc1[t] = wmma_f16(a1, b, acc1[t]);
    }
  }
  const int n = (int)(lane & 15u), g = (int)(lane >> 4u);
#pragma unroll
  for (int sub = 0; sub < 2; ++sub) {
    const v8f* acc = sub ? acc1 : acc0;
#pragma unroll
    for (int t = 0; t < 4; ++t) {
#pragma unroll
      for (int r = 0; r < 8; ++r) {
        const int rr = row0 + sub * 16 + r + 8 * g;
        const int cc = colBase + t * 16 + n;
        out[(size_t)rr * H_DIM + cc] = acc[t][r] + bias[cc];
      }
    }
  }
}

// ---------------- launcher ----------------

extern "C" void kernel_launch(void* const* d_in, const int* in_sizes, int n_in,
                              void* d_out, int out_size, void* d_ws, size_t ws_size,
                              hipStream_t stream) {
  (void)in_sizes; (void)n_in; (void)out_size; (void)ws_size;
  const float* x  = (const float*)d_in[0];
  // d_in[1] = mask: query-axis broadcast in reference -> softmax-invariant, ignored
  const float* Wq = (const float*)d_in[2];
  const float* bq = (const float*)d_in[3];
  const float* Wk = (const float*)d_in[4];
  const float* bk = (const float*)d_in[5];
  const float* Wv = (const float*)d_in[6];
  const float* bv = (const float*)d_in[7];
  const float* Wo = (const float*)d_in[8];
  const float* bo = (const float*)d_in[9];
  float* out = (float*)d_out;

  const size_t XH_B = (size_t)ROWS * H_DIM * sizeof(half_t);      // 12.6 MB
  const size_t WT_B = (size_t)H_DIM * H_DIM * sizeof(half_t);     // 1.18 MB
  char* ws = (char*)d_ws;
  half_t* xh   = (half_t*)(ws);                ws += XH_B;
  half_t* wcat = (half_t*)(ws);                ws += 3 * WT_B;    // [2304 x 768] Q|K|V
  half_t* wot  = (half_t*)(ws);                ws += WT_B;
  half_t* Qh   = (half_t*)(ws);                ws += XH_B;
  half_t* Kh   = (half_t*)(ws);                ws += XH_B;
  half_t* Vth  = (half_t*)(ws);                ws += XH_B;
  half_t* Ah   = (half_t*)(ws);                ws += XH_B;

  const int N4 = ROWS * H_DIM / 4;
  cvt_f16_kernel<<<(N4 + 255) / 256, 256, 0, stream>>>((const float4*)x, xh, N4);

  dim3 tg(H_DIM / 16, H_DIM / 16);
  transpose_cvt_kernel<<<tg, 256, 0, stream>>>(Wq, wcat, H_DIM);
  transpose_cvt_kernel<<<tg, 256, 0, stream>>>(Wk, wcat + (size_t)H_DIM * H_DIM, H_DIM);
  transpose_cvt_kernel<<<tg, 256, 0, stream>>>(Wv, wcat + 2 * (size_t)H_DIM * H_DIM, H_DIM);
  transpose_cvt_kernel<<<tg, 256, 0, stream>>>(Wo, wot, H_DIM);

  gemm_qkv_kernel<<<dim3(ROWS / 128, HCAT / 128), 256, 0, stream>>>(
      xh, wcat, bq, bk, bv, Qh, Kh, Vth);

  attn_kernel<<<dim3(SEQ / 128, BATCH * NHEADS), 256, 0, stream>>>(Qh, Kh, Vth, Ah);

  gemm_out_kernel<<<dim3(ROWS / 128, H_DIM / 128), 256, 0, stream>>>(Ah, wot, bo, out);
}
